// GeneratorCell_41343355191865
// MI455X (gfx1250) — matrix operand order repaired and checked by reading, not compiled, not run on hardware
//
#include <hip/hip_runtime.h>
#include <hip/hip_bf16.h>
#include <math.h>

// Problem constants (match reference)
#define BB 256
#define NN 16354
#define MM 512
#define VV 4096
#define SS 32
#define TM (3*MM)   // 1536

typedef __attribute__((ext_vector_type(16))) __bf16 v16bf;
typedef __attribute__((ext_vector_type(8)))  float  v8f;

__device__ __forceinline__ float sigmoidf_(float x) { return 1.0f / (1.0f + expf(-x)); }

// Load 8 consecutive f32 (16B-aligned) and convert to 8 bf16 fragment elements.
__device__ __forceinline__ void cvt8(const float* __restrict__ p, __bf16* d) {
    const float4* q = (const float4*)p;
    float4 a = q[0], b = q[1];
    d[0] = (__bf16)a.x; d[1] = (__bf16)a.y; d[2] = (__bf16)a.z; d[3] = (__bf16)a.w;
    d[4] = (__bf16)b.x; d[5] = (__bf16)b.y; d[6] = (__bf16)b.z; d[7] = (__bf16)b.w;
}

// ---------------------------------------------------------------------------
// Kernel 1: chained convs -> Ht1, and xt.  One block per batch row, 512 thr.
// ---------------------------------------------------------------------------
__global__ __launch_bounds__(512)
void k1_front(const float* __restrict__ st1, const float* __restrict__ ht1,
              const float* __restrict__ zt,  const int* __restrict__ da_t,
              const float* __restrict__ w5,  const float* __restrict__ b5,
              const float* __restrict__ w12, const float* __restrict__ b12,
              const float* __restrict__ w3,  const float* __restrict__ b3,
              const float* __restrict__ wx,  const float* __restrict__ bx,
              float* __restrict__ Ht1w, float* __restrict__ xtw)
{
    __shared__ float mt[2*MM + 2];          // relu(Mt1) with zero pads, shifted by 1
    const int b = blockIdx.x;
    const int j = threadIdx.x;              // 0..511
    const float* srow = st1 + (size_t)b * NN;

    // c1[i] = sum_k st1[i+k-2]*w5[k] + b5 (zero-pad), at i = 32j and 32j+1
    float c0 = b5[0], c1 = b5[0];
    const int base = SS * j - 2;
#pragma unroll
    for (int k = 0; k < 5; ++k) {
        int i0 = base + k;
        float v0 = (i0 >= 0 && i0 < NN) ? srow[i0] : 0.0f;
        c0 += v0 * w5[k];
        int i1 = i0 + 1;
        float v1 = (i1 >= 0 && i1 < NN) ? srow[i1] : 0.0f;
        c1 += v1 * w5[k];
    }
    float m0 = c0 * w12[0] + c1 * w12[1] + b12[0];

    mt[0] = 0.0f;                            // left zero pad (benign same-value race)
    mt[2*MM + 1] = 0.0f;                     // right zero pad
    mt[1 + j]      = fmaxf(m0, 0.0f);
    mt[1 + MM + j] = fmaxf(ht1[(size_t)b*MM + j], 0.0f);
    __syncthreads();

    // Ht1[j] = relu(Mt1)[2j-1..2j+1] . w3 + b3   (stride 2, pad 1)
    float h = mt[2*j]*w3[0] + mt[2*j + 1]*w3[1] + mt[2*j + 2]*w3[2] + b3[0];
    Ht1w[(size_t)b*MM + j] = h;

    // xt[j] = in[32j]*wx0 + in[32j+1]*wx1 + bx, in = concat(zt, da_f)
    const float* zrow = zt + (size_t)b * (NN - 1);
    float z0 = zrow[SS * j];                                   // always < N-1
    float z1 = (SS*j + 1 < NN - 1) ? zrow[SS*j + 1] : (float)da_t[b];
    xtw[(size_t)b*MM + j] = z0 * wx[0] + z1 * wx[1] + bx[0];
}

// ---------------------------------------------------------------------------
// Kernel 2: gi = xt @ W_ih^T, gh = Ht1 @ W_hh^T via v_wmma_f32_16x16x32_bf16.
// Grid (12,16), 256 thr (8 waves). Wave w -> col tile blockIdx.x*8+w; 16 rows.
// ---------------------------------------------------------------------------
__global__ __launch_bounds__(256)
void k2_gemm(const float* __restrict__ Xw,  const float* __restrict__ Hw,
             const float* __restrict__ Wih, const float* __restrict__ Whh,
             float* __restrict__ gi, float* __restrict__ gh)
{
    const int wave = threadIdx.x >> 5;
    const int lane = threadIdx.x & 31;
    const int colTile = blockIdx.x * 8 + wave;     // 0..95
    const int row0 = blockIdx.y * 16;
    const int col0 = colTile * 16;
    const int m  = lane & 15;                      // A row / B column within tile
    const int hi = lane >> 4;                      // lane half selects K sub-range

    const float* xrow  = Xw  + (size_t)(row0 + m) * MM;
    const float* hrow  = Hw  + (size_t)(row0 + m) * MM;
    const float* wirow = Wih + (size_t)(col0 + m) * MM;   // out col = W row
    const float* whrow = Whh + (size_t)(col0 + m) * MM;

    v8f acc_i = {};
    v8f acc_h = {};
    for (int kc = 0; kc < MM / 32; ++kc) {
        const int kb = kc * 32;
        v16bf ax, ah, bi, bh;
        __bf16* axp = (__bf16*)&ax;
        __bf16* ahp = (__bf16*)&ah;
        __bf16* bip = (__bf16*)&bi;
        __bf16* bhp = (__bf16*)&bh;
        // A 16x32 bf16 layout: elems 0..7 -> K = kb+hi*8+e ; 8..15 -> K = kb+16+hi*8+e
        cvt8(xrow + kb + hi*8,       axp);
        cvt8(xrow + kb + 16 + hi*8,  axp + 8);
        cvt8(hrow + kb + hi*8,       ahp);
        cvt8(hrow + kb + 16 + hi*8,  ahp + 8);
        // B 32x16 bf16 layout: lane=col n, elems 0..7 -> K = kb+hi*8+e ; 8..15 -> K = kb+16+hi*8+e
        cvt8(wirow + kb + hi*8,      bip);
        cvt8(wirow + kb + 16 + hi*8, bip + 8);
        cvt8(whrow + kb + hi*8,      bhp);
        cvt8(whrow + kb + 16 + hi*8, bhp + 8);

        acc_i = __builtin_amdgcn_wmma_f32_16x16x32_bf16(
            false, ax, false, bi, (short)0, acc_i, false, false);
        acc_h = __builtin_amdgcn_wmma_f32_16x16x32_bf16(
            false, ah, false, bh, (short)0, acc_h, false, false);
    }
    // D layout: lane col n = m; VGPR v -> row hi*8 + v
#pragma unroll
    for (int v = 0; v < 8; ++v) {
        int r = hi * 8 + v;
        gi[(size_t)(row0 + r) * TM + col0 + m] = acc_i[v];
        gh[(size_t)(row0 + r) * TM + col0 + m] = acc_h[v];
    }
}

// ---------------------------------------------------------------------------
// Kernel 3: GRU gates -> ht, plus it = floor(4095*sigmoid(ht.w_lht+b)) and ct.
// One block per batch row, 512 thr.
// ---------------------------------------------------------------------------
__global__ __launch_bounds__(512)
void k3_gate(const float* __restrict__ gi, const float* __restrict__ gh,
             const float* __restrict__ Ht1w,
             const float* __restrict__ b_ih, const float* __restrict__ b_hh,
             const float* __restrict__ w_lht, const float* __restrict__ b_lht,
             const float* __restrict__ w_lct, const float* __restrict__ b_lct,
             const int* __restrict__ da_t,
             float* __restrict__ out_ht, int* __restrict__ itw, float* __restrict__ ctw)
{
    __shared__ float r1[MM];
    __shared__ float r2[MM];
    const int b = blockIdx.x;
    const int j = threadIdx.x;
    const size_t o = (size_t)b * TM;

    float ir  = gi[o + j]        + b_ih[j];
    float hr  = gh[o + j]        + b_hh[j];
    float iz  = gi[o + MM + j]   + b_ih[MM + j];
    float hz  = gh[o + MM + j]   + b_hh[MM + j];
    float in_ = gi[o + 2*MM + j] + b_ih[2*MM + j];
    float hn  = gh[o + 2*MM + j] + b_hh[2*MM + j];

    float r  = sigmoidf_(ir + hr);
    float z  = sigmoidf_(iz + hz);
    float n_ = tanhf(in_ + r * hn);
    float hp = Ht1w[(size_t)b*MM + j];
    float ht = (1.0f - z) * n_ + z * hp;
    out_ht[(size_t)b*MM + j] = ht;

    r1[j] = ht * w_lht[j];
    r2[j] = ht * w_lct[j];
    __syncthreads();
    for (int s = MM / 2; s > 0; s >>= 1) {
        if (j < s) { r1[j] += r1[j + s]; r2[j] += r2[j + s]; }
        __syncthreads();
    }
    if (j == 0) {
        int it = (int)floorf((float)(VV - 1) * sigmoidf_(r1[0] + b_lht[0]));
        it = it < 0 ? 0 : (it > VV - 1 ? VV - 1 : it);
        itw[b] = it;
        float da_f = (float)da_t[b];
        ctw[b] = sigmoidf_(r2[0] + da_f * w_lct[MM] + b_lct[0]);
    }
}

// ---------------------------------------------------------------------------
// Kernel 4: st[b,i] = (da+i<N ? st1[b,da+i] : 0)*ct[b] + emb[it[b], i]
// Pure streaming pass; grid (16, 256), 256 thr, 4 elems/thread.
// ---------------------------------------------------------------------------
__global__ __launch_bounds__(256)
void k4_st(const float* __restrict__ st1, const float* __restrict__ emb,
           const int* __restrict__ da_t, const int* __restrict__ itw,
           const float* __restrict__ ctw, float* __restrict__ out_st)
{
    const int b  = blockIdx.y;
    const int i0 = (blockIdx.x * 256 + threadIdx.x) * 4;
    const int da = da_t[b];
    const float ct = ctw[b];
    const float* erow = emb + (size_t)itw[b] * NN;
    const float* srow = st1 + (size_t)b * NN;
    float* orow = out_st + (size_t)b * NN;
#pragma unroll
    for (int e = 0; e < 4; ++e) {
        int i = i0 + e;
        if (i < NN) {
            int idx = da + i;
            float sv = (idx < NN) ? srow[idx] : 0.0f;
            orow[i] = sv * ct + erow[i];
        }
    }
}

// ---------------------------------------------------------------------------
extern "C" void kernel_launch(void* const* d_in, const int* in_sizes, int n_in,
                              void* d_out, int out_size, void* d_ws, size_t ws_size,
                              hipStream_t stream) {
    const float* st1   = (const float*)d_in[0];
    const float* ht1   = (const float*)d_in[1];
    const float* zt    = (const float*)d_in[2];
    const int*   da_t  = (const int*)  d_in[3];
    const float* w_st1 = (const float*)d_in[4];
    const float* b_st1 = (const float*)d_in[5];
    const float* w_st12= (const float*)d_in[6];
    const float* b_st12= (const float*)d_in[7];
    const float* w_mt1 = (const float*)d_in[8];
    const float* b_mt1 = (const float*)d_in[9];
    const float* w_xt  = (const float*)d_in[10];
    const float* b_xt  = (const float*)d_in[11];
    const float* W_ih  = (const float*)d_in[12];
    const float* W_hh  = (const float*)d_in[13];
    const float* b_ih  = (const float*)d_in[14];
    const float* b_hh  = (const float*)d_in[15];
    const float* w_lht = (const float*)d_in[16];
    const float* b_lht = (const float*)d_in[17];
    const float* w_lct = (const float*)d_in[18];
    const float* b_lct = (const float*)d_in[19];
    const float* emb   = (const float*)d_in[20];

    float* out_st = (float*)d_out;                       // [B, N]
    float* out_ht = out_st + (size_t)BB * NN;            // [B, M]

    // Workspace layout (floats): Ht1 | xt | gi | gh | it(int) | ct
    const size_t BM = (size_t)BB * MM;                   // 131072
    float* Ht1w = (float*)d_ws;
    float* xtw  = Ht1w + BM;
    float* giw  = xtw  + BM;
    float* ghw  = giw  + (size_t)BB * TM;
    int*   itw  = (int*)(ghw + (size_t)BB * TM);
    float* ctw  = (float*)(itw + BB);

    // Phase 1: convs -> Ht1, xt
    k1_front<<<dim3(BB), dim3(MM), 0, stream>>>(
        st1, ht1, zt, da_t, w_st1, b_st1, w_st12, b_st12,
        w_mt1, b_mt1, w_xt, b_xt, Ht1w, xtw);

    // Phase 2: WMMA GEMMs gi/gh  (col tiles 96 -> 12 blocks of 8 waves; 16 row tiles)
    k2_gemm<<<dim3(12, 16), dim3(256), 0, stream>>>(xtw, Ht1w, W_ih, W_hh, giw, ghw);

    // Phase 3: gates -> ht, it, ct
    k3_gate<<<dim3(BB), dim3(MM), 0, stream>>>(
        giw, ghw, Ht1w, b_ih, b_hh, w_lht, b_lht, w_lct, b_lct, da_t,
        out_ht, itw, ctw);

    // Phase 4: shifted scale + embedding gather -> st
    k4_st<<<dim3((NN + 1023) / 1024, BB), dim3(256), 0, stream>>>(
        st1, emb, da_t, itw, ctw, out_st);
}
